// GolayConv2d_21303037788553
// MI455X (gfx1250) — matrix operand order repaired and checked by reading, not compiled, not run on hardware
//
#include <hip/hip_runtime.h>

typedef __attribute__((ext_vector_type(16))) __bf16 v16bf;
typedef __attribute__((ext_vector_type(8)))  __bf16 v8bf;
typedef __attribute__((ext_vector_type(8)))  float  v8f;
typedef __attribute__((ext_vector_type(4)))  int    v4i;

typedef __attribute__((address_space(1))) v4i g_v4i;   // global int4
typedef __attribute__((address_space(3))) v4i l_v4i;   // LDS int4

#define B_   32
#define C_   256
#define H_   56
#define W_   56
#define HW_  (H_ * W_)
#define NPIX (B_ * H_ * W_)

// A-pack: [kpos(9)][kt(8)][mt(16)][lane(32)][16 halves] = 1,179,648 bytes
#define APACK_HALVES (9 * 8 * 16 * 32 * 16)
#define APACK_BYTES  (APACK_HALVES * 2)

// conv tiling: 64 consecutive linear pixels per block (HW_ % 64 == 0 ->
// a block never crosses an image). Halo for all 3x3 shifts = +/-57 linear.
#define PIX_PER_BLOCK 64
#define HALO          57
#define STAGE_PIX     (PIX_PER_BLOCK + 2 * HALO)      // 178 pixels staged
#define LDS_STRIDE    264                             // halves/pixel (256+8 pad)
#define LDS_BYTES     (STAGE_PIX * LDS_STRIDE * 2)    // 93,984 B dynamic LDS

#if __has_builtin(__builtin_amdgcn_global_load_async_to_lds_b128)
#define HAVE_ASYNC_COPY 1
#else
#define HAVE_ASYNC_COPY 0
#endif

__device__ __forceinline__ void wait_async_zero() {
#if __has_builtin(__builtin_amdgcn_s_wait_asynccnt)
    __builtin_amdgcn_s_wait_asynccnt(0);
#else
    asm volatile("s_wait_asynccnt 0" ::: "memory");
#endif
}

// ---------------------------------------------------------------------------
// Kernel 1: xt[b][y][x][c] (bf16, NHWC) = FWHT_256( x[b][c][y][x] * golay(c) )
// One wave32 per pixel: 8 channels/lane, 3 in-lane stages + 5 __shfl_xor.
// ---------------------------------------------------------------------------
__global__ void fwht_golay_kernel(const float* __restrict__ x,
                                  __bf16* __restrict__ xt) {
    int gtid = blockIdx.x * blockDim.x + threadIdx.x;
    int pix  = gtid >> 5;
    int lane = threadIdx.x & 31;
    if (pix >= NPIX) return;

    int b  = pix / HW_;
    int hw = pix % HW_;
    const float* px = x + (size_t)b * C_ * HW_ + hw;

    float v[8];
#pragma unroll
    for (int j = 0; j < 8; ++j) {
        int c = lane * 8 + j;
        float val = px[(size_t)c * HW_];
        int s = c & (c >> 1);                       // Rudin-Shapiro sign
        v[j] = (__popc(s) & 1) ? -val : val;
    }
#pragma unroll
    for (int hb = 0; hb < 3; ++hb) {                // bits 0..2 in-lane
        int h = 1 << hb;
#pragma unroll
        for (int j = 0; j < 8; ++j) {
            if ((j & h) == 0) {
                float a = v[j], bb = v[j + h];
                v[j]     = a + bb;
                v[j + h] = a - bb;
            }
        }
    }
#pragma unroll
    for (int mb = 0; mb < 5; ++mb) {                // bits 3..7 cross-lane
        int m = 1 << mb;
        bool upper = (lane & m) != 0;
#pragma unroll
        for (int j = 0; j < 8; ++j) {
            float t = __shfl_xor(v[j], m, 32);
            v[j] = upper ? (t - v[j]) : (v[j] + t);
        }
    }
    v8bf o;
#pragma unroll
    for (int j = 0; j < 8; ++j) o[j] = (__bf16)v[j];
    *(v8bf*)(xt + (size_t)pix * C_ + lane * 8) = o;
}

// ---------------------------------------------------------------------------
// Kernel 2: pack f32 OIHW weights -> bf16 WMMA A-fragments (this file's contiguous k order, matching the B reads):
// lane l holds row M=l&15, half h -> K=h+(l>=16?16:0).
// ---------------------------------------------------------------------------
__global__ void pack_weight_kernel(const float* __restrict__ wgt,
                                   __bf16* __restrict__ apack) {
    int t = blockIdx.x * blockDim.x + threadIdx.x;
    if (t >= 9 * 8 * 16 * 32) return;
    int lane = t & 31;
    int mt   = (t >> 5) & 15;
    int kt   = (t >> 9) & 7;
    int kpos = t >> 12;
    int kh = kpos / 3, kw = kpos % 3;
    int oc   = mt * 16 + (lane & 15);
    int koff = (lane >= 16) ? 16 : 0;

    v16bf frag;
#pragma unroll
    for (int h = 0; h < 16; ++h) {
        int K  = h + koff;
        int ic = kt * 32 + K;
        frag[h] = (__bf16)wgt[(((size_t)oc * C_ + ic) * 3 + kh) * 3 + kw];
    }
    *(v16bf*)(apack + (size_t)t * 16) = frag;
}

// ---------------------------------------------------------------------------
// Kernel 3: implicit-GEMM 3x3 conv via v_wmma_f32_16x16x32_bf16.
// Block: 64 pixels x 256 OCs, activations async-staged to LDS once.
// Wave: 2 N-tiles x 4 M-tiles = 8 accumulators; per kt-step:
// 4 A loads (global, L2-resident) + 2 B reads (LDS) feed 8 WMMAs.
// ---------------------------------------------------------------------------
__global__ void __launch_bounds__(256, 1)
conv_wmma_kernel(const __bf16* __restrict__ xt,
                 const __bf16* __restrict__ apack,
                 const float*  __restrict__ bias,
                 float*        __restrict__ out) {
    extern __shared__ __bf16 lds_xt[];

    const int tid  = threadIdx.x;
    const int lane = tid & 31;
    const int wave = tid >> 5;
    const int pb   = blockIdx.x * PIX_PER_BLOCK;
    const int base = pb - HALO;

    // ---- cooperative stage: xt[base .. base+178) -> LDS, 16B chunks ----
    for (int i = tid; i < STAGE_PIX * 32; i += 256) {
        int p  = base + (i >> 5);
        int c8 = (i & 31) * 8;                            // halves in pixel
        int pc = p < 0 ? 0 : (p >= NPIX ? NPIX - 1 : p);  // clamp (zeroed later)
        const __bf16* src = xt + (size_t)pc * C_ + c8;
        __bf16* dst = lds_xt + (size_t)(i >> 5) * LDS_STRIDE + c8;
#if HAVE_ASYNC_COPY
        __builtin_amdgcn_global_load_async_to_lds_b128(
            (g_v4i*)(void*)src, (l_v4i*)(void*)dst, 0, 0);
#else
        *(v8bf*)dst = *(const v8bf*)src;
#endif
    }
#if HAVE_ASYNC_COPY
    wait_async_zero();
#endif
    __syncthreads();

    const int nsel   = wave >> 2;                 // 0/1: which pixel pair
    const int mbase  = (wave & 3) * 4;            // 4 M-tiles per wave
    const int col    = lane & 15;
    const int kb_off = (lane >= 16) ? 16 : 0;     // B-fragment K offset
    const int bimg   = pb / HW_;                  // whole block in one image

    int yy[2], xx[2], rel[2];
#pragma unroll
    for (int j = 0; j < 2; ++j) {
        int pidx = pb + nsel * 32 + j * 16 + col;
        int hw   = pidx % HW_;
        yy[j]  = hw / W_;
        xx[j]  = hw % W_;
        rel[j] = pidx - base;                     // in [57, 120]
    }

    v8f acc[2][4];
#pragma unroll
    for (int j = 0; j < 2; ++j)
#pragma unroll
        for (int i = 0; i < 4; ++i)
#pragma unroll
            for (int r = 0; r < 8; ++r) acc[j][i][r] = 0.0f;

    v16bf zf;
#pragma unroll
    for (int h = 0; h < 16; ++h) zf[h] = (__bf16)0.0f;

    for (int kpos = 0; kpos < 9; ++kpos) {
        const int dh = kpos / 3 - 1, dw = kpos % 3 - 1;
        const int shift = dh * W_ + dw;           // in [-57, 57]
        bool valid[2];
        int  lbase[2];
#pragma unroll
        for (int j = 0; j < 2; ++j) {
            int ys = yy[j] + dh, xs = xx[j] + dw;
            valid[j] = (ys >= 0) && (ys < H_) && (xs >= 0) && (xs < W_);
            lbase[j] = (rel[j] + shift) * LDS_STRIDE + kb_off; // always in range
        }
#pragma unroll
        for (int kt = 0; kt < 8; ++kt) {
            v16bf bfrag[2];
#pragma unroll
            for (int j = 0; j < 2; ++j) {
                if (valid[j]) bfrag[j] = *(const v16bf*)(lds_xt + lbase[j] + kt * 32);
                else          bfrag[j] = zf;
            }
            const __bf16* ap = apack +
                (((size_t)(kpos * 8 + kt) * 16 + mbase) * 32 + lane) * 16;
#pragma unroll
            for (int i = 0; i < 4; ++i) {
                v16bf afrag = *(const v16bf*)(ap + (size_t)i * 512);
#pragma unroll
                for (int j = 0; j < 2; ++j) {
                    acc[j][i] = __builtin_amdgcn_wmma_f32_16x16x32_bf16(
                        false, afrag, false, bfrag[j], (short)0, acc[j][i],
                        false, false);
                }
            }
        }
    }

    // D layout: VGPR r, lane l -> row r + (l>=16?8:0), col l&15
    const int rowoff = (lane >= 16) ? 8 : 0;
#pragma unroll
    for (int j = 0; j < 2; ++j)
#pragma unroll
        for (int i = 0; i < 4; ++i)
#pragma unroll
            for (int r = 0; r < 8; ++r) {
                int oc = (mbase + i) * 16 + r + rowoff;
                out[((size_t)bimg * C_ + oc) * HW_ + yy[j] * W_ + xx[j]] =
                    acc[j][i][r] + bias[oc];
            }
}

extern "C" void kernel_launch(void* const* d_in, const int* in_sizes, int n_in,
                              void* d_out, int out_size, void* d_ws, size_t ws_size,
                              hipStream_t stream) {
    const float* x    = (const float*)d_in[0];
    const float* wgt  = (const float*)d_in[1];
    const float* bias = (const float*)d_in[2];
    float* out = (float*)d_out;

    __bf16* apack = (__bf16*)d_ws;                        // 1,179,648 B
    __bf16* xt    = (__bf16*)((char*)d_ws + APACK_BYTES); // 51,380,224 B

    hipLaunchKernelGGL(pack_weight_kernel, dim3((9 * 8 * 16 * 32) / 256),
                       dim3(256), 0, stream, wgt, apack);
    hipLaunchKernelGGL(fwht_golay_kernel, dim3(NPIX / 8),
                       dim3(256), 0, stream, x, xt);
    hipLaunchKernelGGL(conv_wmma_kernel, dim3(NPIX / PIX_PER_BLOCK),
                       dim3(256), LDS_BYTES, stream, xt, apack, bias, out);
}
